// MultiHeadAttention_8512625180751
// MI455X (gfx1250) — matrix operand order repaired and checked by reading, not compiled, not run on hardware
//
#include <hip/hip_runtime.h>
#include <stdint.h>
#include <math.h>

// ---------------------------------------------------------------------------
// MultiHeadAttention forward for MI455X (gfx1250, wave32, WMMA)
//   B=2, T=2048, C=1024, H=16, Dh=64, scale = 1/8
// Pipeline:
//   1) f32 -> bf16 conversion of x and the 4 weight matrices; key mask build
//   2) WMMA GEMM (x @ W.T + b) -> Q, K (bf16 row-major), V (bf16 transposed
//      per head: [B][H][Dh][T]); double-buffered async global->LDS staging
//   3) flash-attention (online softmax) with v_wmma_f32_16x16x32_bf16
//   4) WMMA GEMM (attn_out @ Wo.T + bo) -> f32 d_out
// ---------------------------------------------------------------------------

typedef __attribute__((ext_vector_type(16))) __bf16 v16bf;
typedef __attribute__((ext_vector_type(8)))  float  v8f;

#define WMMA_BF16(A, B, Cc) \
  __builtin_amdgcn_wmma_f32_16x16x32_bf16(false, (A), false, (B), (short)0, (Cc), false, false)

static __device__ __forceinline__ unsigned short f2bf(float f) {
  uint32_t u = __float_as_uint(f);
  uint32_t r = (u + 0x7FFFu + ((u >> 16) & 1u)) >> 16;  // round-to-nearest-even
  return (unsigned short)r;
}

// CDNA5 async global->LDS copy, 16B per lane, tracked by ASYNCcnt.
// Low 32 bits of a generic pointer into __shared__ are the LDS byte address
// (ISA 10.2); hardware adds the wave's LDS_BASE.
static __device__ __forceinline__ void async_g2l_b128(void* lds_ptr,
                                                      const void* gptr) {
  uint32_t loff = (uint32_t)(size_t)lds_ptr;
  asm volatile("global_load_async_to_lds_b128 %0, %1, off"
               :: "v"(loff), "v"(gptr)
               : "memory");
}
static __device__ __forceinline__ void wait_async0() {
  asm volatile("s_wait_asynccnt 0" ::: "memory");
}

// Build a 16x32 bf16 fragment from two 16-byte-aligned spans of 8 bf16 each.
static __device__ __forceinline__ v16bf frag_ld(const unsigned short* p0,
                                                const unsigned short* p1) {
  union { v16bf v; uint4 q[2]; } r;
  r.q[0] = *(const uint4*)p0;
  r.q[1] = *(const uint4*)p1;
  return r.v;
}

// ---------------------------------------------------------------------------
__global__ void mha_cvt_bf16(const float* __restrict__ src,
                             unsigned short* __restrict__ dst, int n) {
  int i = blockIdx.x * blockDim.x + threadIdx.x;
  if (i < n) dst[i] = f2bf(src[i]);
}

__global__ void mha_key_mask(const float* __restrict__ x,
                             float* __restrict__ maskAdd, int BT, int C) {
  int i = blockIdx.x * blockDim.x + threadIdx.x;
  if (i < BT) maskAdd[i] = (x[(size_t)i * C] != 0.0f) ? 0.0f : -INFINITY;
}

// ---------------------------------------------------------------------------
// GEMM: out[M][N] = Xb[M][K](bf16) @ Wb[N][K](bf16)^T + bias[N](f32)
// out_mode 0: bf16 row-major [M][N]
// out_mode 1: bf16 V-transpose per head: dst[((b*16+h)*64+d)*2048 + t]
// out_mode 2: f32 row-major [M][N]
// Block: 128 threads (4 waves). Block tile 64x64, K-step 32, double-buffered
// async global->LDS staging.
__global__ __launch_bounds__(128)
void mha_gemm_wmma(const unsigned short* __restrict__ Xb,
                   const unsigned short* __restrict__ Wb,
                   const float* __restrict__ bias,
                   unsigned short* __restrict__ outb,
                   float* __restrict__ outf,
                   int M, int K, int N, int out_mode) {
  __shared__ __align__(16) unsigned short lX[2][64][40];  // 32 cols + 8 pad
  __shared__ __align__(16) unsigned short lW[2][64][40];

  const int tid  = threadIdx.x;
  const int wave = tid >> 5;
  const int lane = tid & 31;
  const int half = lane >> 4;        // 0: lanes 0-15, 1: lanes 16-31
  const int ln   = lane & 15;
  const int n0 = blockIdx.x * 64;
  const int m0 = blockIdx.y * 64;

  const int srow = tid >> 1;         // 0..63
  const int shf  = tid & 1;          // which 16-element chunk of the 32 cols

  auto stage = [&](int buf, int k0) {
    const unsigned short* gx = Xb + (size_t)(m0 + srow) * K + k0 + shf * 16;
    async_g2l_b128(&lX[buf][srow][shf * 16],     gx);
    async_g2l_b128(&lX[buf][srow][shf * 16 + 8], gx + 8);
    const unsigned short* gw = Wb + (size_t)(n0 + srow) * K + k0 + shf * 16;
    async_g2l_b128(&lW[buf][srow][shf * 16],     gw);
    async_g2l_b128(&lW[buf][srow][shf * 16 + 8], gw + 8);
  };

  v8f acc[4] = {};
  stage(0, 0);
  int buf = 0;

  for (int k0 = 0; k0 < K; k0 += 32) {
    wait_async0();       // my fills of `buf` complete
    __syncthreads();     // everyone's fills done; prev reads of buf^1 done
    if (k0 + 32 < K) stage(buf ^ 1, k0 + 32);   // prefetch next tile

    // A fragment: lane row m = ln; 16B at K=half*8 and K=16+half*8
    const int am = wave * 16 + ln;
    v16bf afrag = frag_ld(&lX[buf][am][half * 8], &lX[buf][am][16 + half * 8]);
    v16bf bfr[4];
#pragma unroll
    for (int ct = 0; ct < 4; ++ct) {
      const int bn = ct * 16 + ln;   // B fragment: lane col n, K=half*8 and K=16+half*8
      bfr[ct] = frag_ld(&lW[buf][bn][half * 8], &lW[buf][bn][16 + half * 8]);
    }
#pragma unroll
    for (int ct = 0; ct < 4; ++ct)
      acc[ct] = WMMA_BF16(afrag, bfr[ct], acc[ct]);

    buf ^= 1;
  }

  // ---- epilogue: C/D layout row M = r + half*8, col N = ln ----
#pragma unroll
  for (int ct = 0; ct < 4; ++ct) {
    const int n = n0 + ct * 16 + ln;
    const float bs = bias[n];
#pragma unroll
    for (int r = 0; r < 8; ++r) {
      const int m = m0 + wave * 16 + r + half * 8;
      const float val = acc[ct][r] + bs;
      if (out_mode == 0) {
        outb[(size_t)m * N + n] = f2bf(val);
      } else if (out_mode == 1) {
        const int b = m >> 11, t = m & 2047;     // T = 2048
        const int h = n >> 6,  d = n & 63;       // Dh = 64
        outb[(((size_t)b * 16 + h) * 64 + d) * 2048 + t] = f2bf(val);
      } else {
        outf[(size_t)m * N + n] = val;
      }
    }
  }
}

// ---------------------------------------------------------------------------
// Flash attention. Grid: (T/64, H, B). Block: 128 threads (4 waves).
// Each wave owns 16 query rows; Tk streamed in chunks of 32 with
// double-buffered async global->LDS staging of K and V^T tiles.
__global__ __launch_bounds__(128)
void mha_flash_attn(const unsigned short* __restrict__ Qb,   // [B*T][C] bf16
                    const unsigned short* __restrict__ Kb,   // [B*T][C] bf16
                    const unsigned short* __restrict__ Vtb,  // [B][H][Dh][T] bf16
                    const float* __restrict__ maskAdd,       // [B*T]
                    unsigned short* __restrict__ Ob) {       // [B*T][C] bf16
  const int T = 2048, C = 1024, H = 16;
  const float scale = 0.125f;  // 1/sqrt(64)

  __shared__ __align__(16) unsigned short lK[2][32][72];    // [tk][d] +8 pad
  __shared__ __align__(16) unsigned short lV[2][64][40];    // [d][tk] +8 pad
  __shared__ __align__(16) unsigned short lP[4][16][40];    // per-wave P tile

  const int tid  = threadIdx.x;
  const int wave = tid >> 5;
  const int lane = tid & 31;
  const int half = lane >> 4;
  const int ln   = lane & 15;

  const int tq0 = blockIdx.x * 64 + wave * 16;
  const int hh  = blockIdx.y;
  const int bb  = blockIdx.z;

  auto stage = [&](int buf, int tk0) {
    const int kr = tid >> 2, kq = tid & 3;   // 32 rows x 4 chunks of 16
    const unsigned short* gk =
        Kb + (size_t)(bb * T + tk0 + kr) * C + hh * 64 + kq * 16;
    async_g2l_b128(&lK[buf][kr][kq * 16],     gk);
    async_g2l_b128(&lK[buf][kr][kq * 16 + 8], gk + 8);
    const int vr = tid >> 1, vh = tid & 1;   // 64 rows x 2 chunks of 16
    const unsigned short* gv =
        Vtb + (((size_t)bb * H + hh) * 64 + vr) * T + tk0 + vh * 16;
    async_g2l_b128(&lV[buf][vr][vh * 16],     gv);
    async_g2l_b128(&lV[buf][vr][vh * 16 + 8], gv + 8);
  };

  // Q fragments for this wave's 16 rows (Dh = 64 -> two 16x32 A-frags)
  const unsigned short* qrow = Qb + ((size_t)(bb * T + tq0 + ln)) * C + hh * 64;
  v16bf qf0 = frag_ld(qrow + half * 8,      qrow + 16 + half * 8);
  v16bf qf1 = frag_ld(qrow + 32 + half * 8, qrow + 48 + half * 8);

  v8f oacc[4] = {};
  float mrow[8], lrow[8];
#pragma unroll
  for (int r = 0; r < 8; ++r) { mrow[r] = -INFINITY; lrow[r] = 0.0f; }

  stage(0, 0);
  int buf = 0;

  for (int tk0 = 0; tk0 < T; tk0 += 32) {
    wait_async0();
    __syncthreads();
    if (tk0 + 32 < T) stage(buf ^ 1, tk0 + 32);   // prefetch next chunk

    // ---- scores: two 16x16 tiles (tk sub-cols 0-15, 16-31), 4 WMMAs ----
    v16bf k00 = frag_ld(&lK[buf][ln][half * 8],           &lK[buf][ln][16 + half * 8]);
    v16bf k01 = frag_ld(&lK[buf][ln][32 + half * 8],      &lK[buf][ln][48 + half * 8]);
    v16bf k10 = frag_ld(&lK[buf][16 + ln][half * 8],      &lK[buf][16 + ln][16 + half * 8]);
    v16bf k11 = frag_ld(&lK[buf][16 + ln][32 + half * 8], &lK[buf][16 + ln][48 + half * 8]);
    v8f s0 = {}, s1 = {};
    s0 = WMMA_BF16(qf0, k00, s0);
    s1 = WMMA_BF16(qf0, k10, s1);
    s0 = WMMA_BF16(qf1, k01, s0);
    s1 = WMMA_BF16(qf1, k11, s1);

    const float madd0 = maskAdd[bb * T + tk0 + ln];
    const float madd1 = maskAdd[bb * T + tk0 + 16 + ln];
#pragma unroll
    for (int r = 0; r < 8; ++r) {
      s0[r] = s0[r] * scale + madd0;
      s1[r] = s1[r] * scale + madd1;
    }

    // ---- online softmax (row = r + half*8; row spans 16 lanes of a half) --
#pragma unroll
    for (int r = 0; r < 8; ++r) {
      float mx = fmaxf(s0[r], s1[r]);
#pragma unroll
      for (int off = 1; off < 16; off <<= 1)
        mx = fmaxf(mx, __shfl_xor(mx, off, 32));
      mx = fmaxf(mx, -1.0e30f);            // keep finite if whole chunk masked
      const float mn = fmaxf(mrow[r], mx);
      const float alpha = __expf(mrow[r] - mn);
      const float p0 = __expf(s0[r] - mn);
      const float p1 = __expf(s1[r] - mn);
      float sum = p0 + p1;
#pragma unroll
      for (int off = 1; off < 16; off <<= 1)
        sum += __shfl_xor(sum, off, 32);
      lrow[r] = lrow[r] * alpha + sum;
      mrow[r] = mn;
#pragma unroll
      for (int dt = 0; dt < 4; ++dt) oacc[dt][r] *= alpha;
      // C-layout -> LDS (P tile, bf16)
      lP[wave][r + half * 8][ln]      = f2bf(p0);
      lP[wave][r + half * 8][16 + ln] = f2bf(p1);
    }

    // ---- P (16x32) @ V (32x64): per-wave LDS round-trip; DS ops are
    //      in-order within a wave ----
    v16bf pf = frag_ld(&lP[wave][ln][half * 8], &lP[wave][ln][16 + half * 8]);
    v16bf vf[4];
#pragma unroll
    for (int dt = 0; dt < 4; ++dt) {
      const int bn = dt * 16 + ln;
      vf[dt] = frag_ld(&lV[buf][bn][half * 8], &lV[buf][bn][16 + half * 8]);
    }
#pragma unroll
    for (int dt = 0; dt < 4; ++dt)
      oacc[dt] = WMMA_BF16(pf, vf[dt], oacc[dt]);

    buf ^= 1;
  }

  // ---- epilogue: normalize and store bf16 attn output [B*T][C] ----
#pragma unroll
  for (int r = 0; r < 8; ++r) {
    const float inv = 1.0f / lrow[r];
    const int t = tq0 + r + half * 8;
#pragma unroll
    for (int dt = 0; dt < 4; ++dt) {
      Ob[((size_t)(bb * T + t)) * C + hh * 64 + dt * 16 + ln] =
          f2bf(oacc[dt][r] * inv);
    }
  }
}

// ---------------------------------------------------------------------------
extern "C" void kernel_launch(void* const* d_in, const int* in_sizes, int n_in,
                              void* d_out, int out_size, void* d_ws, size_t ws_size,
                              hipStream_t stream) {
  (void)in_sizes; (void)n_in; (void)out_size; (void)ws_size;

  const float* x    = (const float*)d_in[0];
  const float* wq_w = (const float*)d_in[1];
  const float* wq_b = (const float*)d_in[2];
  const float* wk_w = (const float*)d_in[3];
  const float* wk_b = (const float*)d_in[4];
  const float* wv_w = (const float*)d_in[5];
  const float* wv_b = (const float*)d_in[6];
  const float* wo_w = (const float*)d_in[7];
  const float* wo_b = (const float*)d_in[8];

  const int B = 2, T = 2048, C = 1024, H = 16;
  const int BT = B * T;                    // 4096
  const size_t BTC = (size_t)BT * C;       // 4,194,304
  const size_t CC  = (size_t)C * C;        // 1,048,576

  // workspace carve-out (~48 MB total)
  char* ws = (char*)d_ws;
  size_t off = 0;
  auto carve = [&](size_t bytes) {
    size_t o = off;
    off += (bytes + 255) & ~(size_t)255;
    return o;
  };
  unsigned short* xb  = (unsigned short*)(ws + carve(BTC * 2));
  unsigned short* qb  = (unsigned short*)(ws + carve(BTC * 2));
  unsigned short* kb  = (unsigned short*)(ws + carve(BTC * 2));
  unsigned short* vtb = (unsigned short*)(ws + carve(BTC * 2));  // [B][H][Dh][T]
  unsigned short* ab  = (unsigned short*)(ws + carve(BTC * 2));
  unsigned short* wqb = (unsigned short*)(ws + carve(CC * 2));
  unsigned short* wkb = (unsigned short*)(ws + carve(CC * 2));
  unsigned short* wvb = (unsigned short*)(ws + carve(CC * 2));
  unsigned short* wob = (unsigned short*)(ws + carve(CC * 2));
  float* maskp        = (float*)(ws + carve((size_t)BT * 4));

  // 1) conversions + mask
  mha_cvt_bf16<<<(int)((BTC + 255) / 256), 256, 0, stream>>>(x, xb, (int)BTC);
  mha_cvt_bf16<<<(int)((CC + 255) / 256), 256, 0, stream>>>(wq_w, wqb, (int)CC);
  mha_cvt_bf16<<<(int)((CC + 255) / 256), 256, 0, stream>>>(wk_w, wkb, (int)CC);
  mha_cvt_bf16<<<(int)((CC + 255) / 256), 256, 0, stream>>>(wv_w, wvb, (int)CC);
  mha_cvt_bf16<<<(int)((CC + 255) / 256), 256, 0, stream>>>(wo_w, wob, (int)CC);
  mha_key_mask<<<(BT + 255) / 256, 256, 0, stream>>>(x, maskp, BT, C);

  // 2) Q, K, V projections
  dim3 gg(C / 64, BT / 64);   // (16, 64)
  mha_gemm_wmma<<<gg, 128, 0, stream>>>(xb, wqb, wq_b, qb, nullptr, BT, C, C, 0);
  mha_gemm_wmma<<<gg, 128, 0, stream>>>(xb, wkb, wk_b, kb, nullptr, BT, C, C, 0);
  mha_gemm_wmma<<<gg, 128, 0, stream>>>(xb, wvb, wv_b, vtb, nullptr, BT, C, C, 1);

  // 3) attention
  dim3 ga(T / 64, H, B);      // (32, 16, 2)
  mha_flash_attn<<<ga, 128, 0, stream>>>(qb, kb, vtb, maskp, ab);

  // 4) output projection -> f32 d_out
  mha_gemm_wmma<<<gg, 128, 0, stream>>>(ab, wob, wo_b, nullptr, (float*)d_out,
                                        BT, C, C, 2);
}
